// WaveFinderKAN_23957327577240
// MI455X (gfx1250) — matrix operand order repaired and checked by reading, not compiled, not run on hardware
//
#include <hip/hip_runtime.h>
#include <hip/hip_bf16.h>

typedef _Float16 h16 __attribute__((ext_vector_type(16)));
typedef _Float16 h8  __attribute__((ext_vector_type(8)));
typedef _Float16 h2  __attribute__((ext_vector_type(2)));
typedef float    v8f __attribute__((ext_vector_type(8)));

#define WAVES_PER_BLOCK 4
#define MTILES 2                      // 2 x 16 = 32 batch rows per wave
// -200 * log2(e): exp(-200*d^2) == exp2(C * d^2)
#define EXP2_SCALE (-288.5390081777927f)

// ---------------------------------------------------------------------------
// Prepass: spline (4096 x 32, f32, row-major) -> f16 in d_ws, pre-swizzled
// into the wave32 WMMA B-matrix layout for 16x16x32 f16:
//   B tile (K=32, N=16): lanes 0-15 hold column N=lane, elements e -> K=e (e<8), K=16+e-8 (e>=8);
//                        lanes 16-31 hold column N=lane-16, elements e -> K=8+e (e<8), K=24+e-8 (e>=8).
// Flat layout: ws[((kb*2 + nb)*32 + lane)*16 + e]
// ---------------------------------------------------------------------------
__global__ void kan_prep_spline_f16(const float* __restrict__ spline,
                                    _Float16* __restrict__ wsB) {
    int t = blockIdx.x * blockDim.x + threadIdx.x;   // one thread per f16 elem
    int e    = t & 15;
    int lane = (t >> 4) & 31;
    int nb   = (t >> 9) & 1;
    int kb   = t >> 10;
    int n = nb * 16 + (lane & 15);
    int k = kb * 32 + ((e < 8) ? ((lane >> 4) * 8 + e) : (16 + (lane >> 4) * 8 + (e - 8)));
    wsB[t] = (_Float16)spline[k * 32 + n];
}

// ---------------------------------------------------------------------------
// Main kernel: one wave = two 16-row batch tiles (32 rows).
//   Phase 1: separable RBF factors ex[64], ey[64] per row -> LDS (f16).
//   Phase 2: K-loop over 4096 grid points in steps of 32; per iteration two
//            A tiles are built as exi * ey[j] in WMMA A-register layout and
//            each loaded B tile feeds two v_wmma_f32_16x16x32_f16 (B reuse
//            across M halves the L2 traffic; the two A constructions give
//            independent VALU work to hide WMMA hazards).
//   Epilogue: + x@base_w^T, tanh, tiny MLP head with wave32 shfl reduction.
// ---------------------------------------------------------------------------
__global__ void __launch_bounds__(WAVES_PER_BLOCK * 32)
kan_main(const float* __restrict__ x,
         const float* __restrict__ base_w,
         const float* __restrict__ w1,
         const float* __restrict__ b1,
         const float* __restrict__ w2,
         const float* __restrict__ b2,
         const _Float16* __restrict__ wsB,
         float* __restrict__ out) {
    __shared__ __align__(16) _Float16 s_exy[WAVES_PER_BLOCK][MTILES][16][128];
    __shared__ float s_h[WAVES_PER_BLOCK][MTILES][16][32];

    const int lane = threadIdx.x & 31;
    const int wave = threadIdx.x >> 5;
    const int r    = lane & 15;   // local batch row handled by this lane
    const int hi   = lane >> 4;   // 0: ex / low-K half, 1: ey / high-K half
    const int rowbase = (blockIdx.x * WAVES_PER_BLOCK + wave) * (16 * MTILES);

    // ---- Phase 1: separable RBF factors (128 exp2's per lane) -----------
    #pragma unroll
    for (int t = 0; t < MTILES; ++t) {
        const float xv = x[2 * (rowbase + t * 16 + r) + hi];  // x0->ex, x1->ey
        #pragma unroll 4
        for (int i = 0; i < 64; i += 2) {
            float g0 = -1.0f + (float)i       * (2.0f / 63.0f);
            float g1 = -1.0f + (float)(i + 1) * (2.0f / 63.0f);
            float d0 = xv - g0;
            float d1 = xv - g1;
            h2 pk;
            pk[0] = (_Float16)__builtin_amdgcn_exp2f(d0 * d0 * EXP2_SCALE);
            pk[1] = (_Float16)__builtin_amdgcn_exp2f(d1 * d1 * EXP2_SCALE);
            *(h2*)&s_exy[wave][t][r][hi * 64 + i] = pk;
        }
    }
    __syncthreads();

    // ---- Phase 2: WMMA K-loop ------------------------------------------
    v8f c00 = {};   // tile 0, features  0..15
    v8f c01 = {};   // tile 0, features 16..31
    v8f c10 = {};   // tile 1, features  0..15
    v8f c11 = {};   // tile 1, features 16..31
    const int off = hi ? 8 : 0;   // A-layout K-half offset within j
    const h16* __restrict__ Bt = (const h16*)wsB;

    for (int kb = 0; kb < 128; ++kb) {
        const int i  = kb >> 1;         // grid x index (constant in 32-block)
        const int jb = (kb & 1) << 5;   // grid y block base (0 or 32)

        const _Float16 exi0 = s_exy[wave][0][r][i];
        const _Float16 exi1 = s_exy[wave][1][r][i];
        const h8 ey00 = *(const h8*)&s_exy[wave][0][r][64 + jb + off];
        const h8 ey01 = *(const h8*)&s_exy[wave][0][r][64 + jb + off + 16];
        const h8 ey10 = *(const h8*)&s_exy[wave][1][r][64 + jb + off];
        const h8 ey11 = *(const h8*)&s_exy[wave][1][r][64 + jb + off + 16];

        // A tiles in WMMA register layout: lane<16: K=e / K=16+e halves,
        // lane>=16: K=8+e / K=24+e halves (off handles the shift).
        h16 a0, a1;
        #pragma unroll
        for (int e = 0; e < 8; ++e) {
            a0[e]     = exi0 * ey00[e];
            a0[8 + e] = exi0 * ey01[e];
            a1[e]     = exi1 * ey10[e];
            a1[8 + e] = exi1 * ey11[e];
        }

        const h16 bA = Bt[(kb * 2 + 0) * 32 + lane];
        const h16 bB = Bt[(kb * 2 + 1) * 32 + lane];

        c00 = __builtin_amdgcn_wmma_f32_16x16x32_f16(
                false, a0, false, bA, (short)0, c00, false, false);
        c10 = __builtin_amdgcn_wmma_f32_16x16x32_f16(
                false, a1, false, bA, (short)0, c10, false, false);
        c01 = __builtin_amdgcn_wmma_f32_16x16x32_f16(
                false, a0, false, bB, (short)0, c01, false, false);
        c11 = __builtin_amdgcn_wmma_f32_16x16x32_f16(
                false, a1, false, bB, (short)0, c11, false, false);
    }

    // ---- Epilogue: base linear + tanh, staged to LDS -------------------
    // C/D layout: lane<16: N=lane, VGPR q -> M=q; lane>=16: N=lane-16, M=8+q.
    const int nA = r;
    const int nB = 16 + r;
    const float bwA0 = base_w[2 * nA + 0], bwA1 = base_w[2 * nA + 1];
    const float bwB0 = base_w[2 * nB + 0], bwB1 = base_w[2 * nB + 1];
    #pragma unroll
    for (int t = 0; t < MTILES; ++t) {
        const v8f cAa = t ? c10 : c00;
        const v8f cBb = t ? c11 : c01;
        #pragma unroll
        for (int q = 0; q < 8; ++q) {
            const int m = hi ? (8 + q) : q;
            const int row = rowbase + t * 16 + m;
            const float x0 = x[2 * row + 0];
            const float x1 = x[2 * row + 1];
            const float hA = cAa[q] + x0 * bwA0 + x1 * bwA1;
            const float hB = cBb[q] + x0 * bwB0 + x1 * bwB1;
            s_h[wave][t][m][nA] = tanhf(hA);
            s_h[wave][t][m][nB] = tanhf(hB);
        }
    }
    __syncthreads();

    // ---- MLP head: lane = hidden unit, wave32 shuffle reduction --------
    float w1r[32];
    #pragma unroll
    for (int k2 = 0; k2 < 32; ++k2) w1r[k2] = w1[lane * 32 + k2];
    const float b1v = b1[lane];
    const float w2v = w2[lane];
    const float b2v = b2[0];

    #pragma unroll
    for (int t = 0; t < MTILES; ++t) {
        for (int m = 0; m < 16; ++m) {
            float g = b1v;
            #pragma unroll
            for (int k2 = 0; k2 < 32; ++k2) g += s_h[wave][t][m][k2] * w1r[k2];
            float v = tanhf(g) * w2v;
            #pragma unroll
            for (int ofs = 16; ofs >= 1; ofs >>= 1) v += __shfl_xor(v, ofs, 32);
            if (lane == 0) out[rowbase + t * 16 + m] = v + b2v;
        }
    }
}

// ---------------------------------------------------------------------------
// setup_inputs order: x, grid, spline, base_w, w1, b1, w2, b2
// (grid is recomputed analytically; d_in[1] unused)
// ---------------------------------------------------------------------------
extern "C" void kernel_launch(void* const* d_in, const int* in_sizes, int n_in,
                              void* d_out, int out_size, void* d_ws, size_t ws_size,
                              hipStream_t stream) {
    const float* x      = (const float*)d_in[0];
    const float* spline = (const float*)d_in[2];
    const float* base_w = (const float*)d_in[3];
    const float* w1     = (const float*)d_in[4];
    const float* b1     = (const float*)d_in[5];
    const float* w2     = (const float*)d_in[6];
    const float* b2     = (const float*)d_in[7];
    float* out = (float*)d_out;
    _Float16* wsB = (_Float16*)d_ws;   // 4096*32 f16 = 256 KB swizzled spline

    const int splineElems = in_sizes[2];             // 131072
    const int batch       = in_sizes[0] / 2;         // 131072

    kan_prep_spline_f16<<<splineElems / 256, 256, 0, stream>>>(spline, wsB);

    const int blocks = batch / (16 * MTILES * WAVES_PER_BLOCK);  // 1024
    kan_main<<<blocks, WAVES_PER_BLOCK * 32, 0, stream>>>(
        x, base_w, w1, b1, w2, b2, wsB, out);
}
